// CustomLMHead_55482387529894
// MI455X (gfx1250) — matrix operand order repaired and checked by reading, not compiled, not run on hardware
//
#include <hip/hip_runtime.h>

// ---------------------------------------------------------------------------
// LM head: out[2048,151936] = x[2048,896] @ W^T, W = int8-in-int32 * row scale
// bf16 WMMA (int8 weights exact in bf16), f32 accumulate, scale in epilogue.
// Workgroup owns 128 weight rows staged once into LDS (~231 KB; W read from
// HBM exactly once). 16 waves: wave&1 selects a 64-column half, wave>>1 the
// M-group; each wave runs 4 M-tiles x 4 subtiles per pass so every B fragment
// from LDS feeds 4 WMMAs (0.5 ds_load_b128 per WMMA), 16 WMMAs per K-step.
// NTILE=128 halves workgroup count vs 64 -> x L2 re-read traffic ~4.35 GB.
// ---------------------------------------------------------------------------

#define IN_F    896
#define OUT_F   151936
#define M_TOT   2048
#define KSTEPS  (IN_F / 32)      // 28
#define NTILE   128              // weight rows per workgroup
#define LDS_STRIDE (IN_F + 8)    // 904 bf16 per row: +8 pad to spread LDS banks
#define BLOCK   512              // 16 waves (wave32)
#define MWAVE   4                // M-tiles per wave iteration

typedef __attribute__((ext_vector_type(16))) __bf16 v16bf;
typedef __attribute__((ext_vector_type(8)))  __bf16 v8bf;
typedef __attribute__((ext_vector_type(4)))  __bf16 v4bf;
typedef __attribute__((ext_vector_type(8)))  float  v8f;
typedef __attribute__((ext_vector_type(4)))  float  v4f;
typedef __attribute__((ext_vector_type(4)))  int    v4i;

// ---- pre-pass: x fp32 -> bf16 (row-major [2048][896]) into workspace ------
__global__ void cvt_x_kernel(const float* __restrict__ x,
                             __bf16* __restrict__ xb, int n4) {
    int i = blockIdx.x * blockDim.x + threadIdx.x;
    if (i < n4) {
        v4f v = ((const v4f*)x)[i];
        v4bf o;
        o[0] = (__bf16)v[0]; o[1] = (__bf16)v[1];
        o[2] = (__bf16)v[2]; o[3] = (__bf16)v[3];
        ((v4bf*)xb)[i] = o;
    }
}

// ---- main GEMM ------------------------------------------------------------
__global__ void __launch_bounds__(BLOCK)
lmhead_wmma_kernel(const __bf16* __restrict__ xb,
                   const int*    __restrict__ qw,
                   const float*  __restrict__ scales,
                   float*        __restrict__ out) {
    extern __shared__ __bf16 lds[];          // NTILE * LDS_STRIDE bf16 (~231 KB)

    const int n0 = blockIdx.x * NTILE;

    // ---- stage 128 weight rows: int32 -> bf16 into LDS (read W once) ----
    {
        const int* wt = qw + (size_t)n0 * IN_F;
        const int total4 = (NTILE * IN_F) / 4;         // 28672, exact /BLOCK
        for (int idx = threadIdx.x; idx < total4; idx += BLOCK) {
            v4i w4 = ((const v4i*)wt)[idx];            // coalesced b128 loads
            int e   = idx * 4;
            int row = e / IN_F;
            int k   = e - row * IN_F;
            v4bf b;
            b[0] = (__bf16)(float)w4[0];
            b[1] = (__bf16)(float)w4[1];
            b[2] = (__bf16)(float)w4[2];
            b[3] = (__bf16)(float)w4[3];
            *(v4bf*)&lds[row * LDS_STRIDE + k] = b;    // 8B aligned
        }
    }
    __syncthreads();

    const int lane   = threadIdx.x & 31;
    const int wave   = threadIdx.x >> 5;   // 0..15
    const int lr     = lane & 15;
    const int hi     = lane >> 4;          // 0/1 half-wave
    const int colgrp = wave & 1;           // 64-column half of the 128-row tile
    const int mgrp   = wave >> 1;          // 0..7 M-group

    const __bf16* bb = &lds[(size_t)(colgrp * 64 + lr) * LDS_STRIDE + hi * 8];

    // 8 M-groups x MWAVE(4) M-tiles per pass; 128/(8*4) = 4 passes -> M=2048.
    #pragma unroll 1
    for (int grp = 0; grp < (M_TOT / 16) / (8 * MWAVE); ++grp) {
        const int mt = (grp * 8 + mgrp) * MWAVE;
        const int m0 = mt * 16;

        v8f acc[MWAVE][4];
        const v8f vz = {0.f,0.f,0.f,0.f,0.f,0.f,0.f,0.f};
        #pragma unroll
        for (int j = 0; j < MWAVE; ++j)
            for (int s = 0; s < 4; ++s) acc[j][s] = vz;

        // A-fragment base: lane lr -> row m0+lr; half-wave -> K offset 0/8;
        // M-tile j is a compile-time immediate offset j*16*IN_F.
        const __bf16* arow = xb + (size_t)(m0 + lr) * IN_F + hi * 8;

        #pragma unroll 1
        for (int kk = 0; kk < KSTEPS; ++kk) {
            const int k0 = kk * 32;

            v16bf a[MWAVE];
            #pragma unroll
            for (int j = 0; j < MWAVE; ++j) {
                const __bf16* ap = arow + (size_t)j * 16 * IN_F + k0;
                v8bf a0 = *(const v8bf*)(ap);          // K k0..k0+7  (|+8 hi)
                v8bf a1 = *(const v8bf*)(ap + 16);     // K k0+16..23 (|+8 hi)
                a[j] = __builtin_shufflevector(a0, a1,
                           0,1,2,3,4,5,6,7,8,9,10,11,12,13,14,15);
            }

            #pragma unroll
            for (int sub = 0; sub < 4; ++sub) {
                const __bf16* bp = bb + (size_t)(sub * 16) * LDS_STRIDE + k0;
                v8bf b0 = *(const v8bf*)(bp);          // ds_load_b128
                v8bf b1 = *(const v8bf*)(bp + 16);     // ds_load_b128
                v16bf b = __builtin_shufflevector(b0, b1,
                              0,1,2,3,4,5,6,7,8,9,10,11,12,13,14,15);
                #pragma unroll
                for (int j = 0; j < MWAVE; ++j)        // reuse B across 4 tiles
                    acc[j][sub] = __builtin_amdgcn_wmma_f32_16x16x32_bf16(
                                      false, a[j], false, b, (short)0,
                                      acc[j][sub], false, false);
            }
        }

        // ---- epilogue: scale by per-output-column scale, store f32 ----
        #pragma unroll
        for (int j = 0; j < MWAVE; ++j) {
            #pragma unroll
            for (int sub = 0; sub < 4; ++sub) {
                const int   n = n0 + colgrp * 64 + sub * 16 + lr;
                const float s = scales[n];
                float* orow = out + (size_t)(m0 + j * 16 + hi * 8) * OUT_F + n;
                #pragma unroll
                for (int r = 0; r < 8; ++r)
                    orow[(size_t)r * OUT_F] = acc[j][sub][r] * s;
            }
        }
    }
}

extern "C" void kernel_launch(void* const* d_in, const int* in_sizes, int n_in,
                              void* d_out, int out_size, void* d_ws, size_t ws_size,
                              hipStream_t stream) {
    const float* x      = (const float*)d_in[0];
    const int*   qw     = (const int*)  d_in[1];
    const float* scales = (const float*)d_in[2];
    float*       out    = (float*)d_out;
    __bf16*      xb     = (__bf16*)d_ws;        // 2048*896*2 = 3.67 MB

    const int n4 = (M_TOT * IN_F) / 4;          // 458752
    cvt_x_kernel<<<(n4 + 255) / 256, 256, 0, stream>>>(x, xb, n4);

    const size_t ldsBytes = (size_t)NTILE * LDS_STRIDE * sizeof(__bf16); // 231424
    // allow >64KB dynamic LDS (320KB/WGP on CDNA5); host-side property change,
    // not a stream op, so graph-capture safe.
    hipFuncSetAttribute((const void*)lmhead_wmma_kernel,
                        hipFuncAttributeMaxDynamicSharedMemorySize,
                        (int)ldsBytes);

    lmhead_wmma_kernel<<<OUT_F / NTILE, BLOCK, ldsBytes, stream>>>(
        xb, qw, scales, out);
}
